// CausalSelfAttention_3607772529016
// MI455X (gfx1250) — matrix operand order repaired and checked by reading, not compiled, not run on hardware
//
#include <hip/hip_runtime.h>

// ---------------------------------------------------------------------------
// CDNA5 (gfx1250) causal self-attention: bf16 WMMA (f32 accumulate) pipeline
//   1) f32 -> bf16 conversion of x, Wqkv, Wo
//   2) QKV projection GEMM  (v_wmma_f32_16x16x32_bf16, double-buffered LDS)
//   3) flash-attention      (v_wmma_f32_16x16x32_bf16, online softmax,
//                            async global->LDS V staging, prefetch)
//   4) output projection GEMM -> f32
// ---------------------------------------------------------------------------

typedef __attribute__((ext_vector_type(16))) __bf16 v16bf;
typedef __attribute__((ext_vector_type(8)))  __bf16 v8bf;
typedef __attribute__((ext_vector_type(8)))  float  v8f;
typedef __attribute__((ext_vector_type(4)))  int    v4i;

#define DMODEL 1024
#define NHEADS 16
#define DHEAD  64
#define BATCH  2
#define SEQ    2048
#define ROWS   (BATCH * SEQ)   // 4096

#if defined(__gfx1250__) && \
    __has_builtin(__builtin_amdgcn_global_load_async_to_lds_b128) && \
    __has_builtin(__builtin_amdgcn_s_wait_asynccnt)
#define USE_ASYNC_LDS 1
#else
#define USE_ASYNC_LDS 0
#endif

__device__ __forceinline__ v8f vzero8() {
  v8f z;
#pragma unroll
  for (int i = 0; i < 8; ++i) z[i] = 0.0f;
  return z;
}

__device__ __forceinline__ v16bf pack16(v8bf lo, v8bf hi) {
  v16bf v;
#pragma unroll
  for (int i = 0; i < 8; ++i) { v[i] = lo[i]; v[i + 8] = hi[i]; }
  return v;
}

__device__ __forceinline__ v8f wmma_bf16(v16bf a, v16bf b, v8f c) {
  // D = A(16x32 bf16) * B(32x16 bf16) + C(16x16 f32)
  return __builtin_amdgcn_wmma_f32_16x16x32_bf16(false, a, false, b,
                                                 (short)0, c, false, false);
}

// ---------------------------------------------------------------------------
// f32 -> bf16 conversion (grid-stride)
// ---------------------------------------------------------------------------
__global__ void cvt_f32_to_bf16(const float* __restrict__ src,
                                __bf16* __restrict__ dst, int n) {
  int i = blockIdx.x * blockDim.x + threadIdx.x;
  int stride = gridDim.x * blockDim.x;
  for (; i < n; i += stride) dst[i] = (__bf16)src[i];
}

// ---------------------------------------------------------------------------
// Tiled GEMM: C[M,N] = A[M,K] * B[N,K]^T   (A,B bf16; accumulate f32)
// Block: 256 threads = 8 waves (2x4), block tile 128(M) x 128(N), K-step 32.
// Each wave computes 64x32 = 4x2 WMMA accumulators (8 WMMA / K-step).
// Double-buffered LDS: next K-step's global loads are issued before the
// WMMA chain so L2 latency hides under the matrix ops.
// ---------------------------------------------------------------------------
template <bool OUT_BF16>
__global__ __launch_bounds__(256)
void gemm_bf16(const __bf16* __restrict__ A, const __bf16* __restrict__ B,
               float* __restrict__ Cf, __bf16* __restrict__ Cb,
               int M, int N, int K) {
  __shared__ __bf16 As[2][128 * 40];    // [row][k]  pitch 40 (pad 8)
  __shared__ __bf16 Bt[2][32 * 136];    // [k][n]    pitch 136 (pad 8)

  const int t    = threadIdx.x;
  const int wid  = t >> 5;
  const int lane = t & 31;
  const int half = lane >> 4;
  const int l16  = lane & 15;
  const int wm   = wid >> 2;            // 0..1
  const int wn   = wid & 3;             // 0..3
  const int m0   = blockIdx.y * 128;
  const int n0   = blockIdx.x * 128;

  // staging assignment: each thread moves 16 bf16 of A and 16 bf16 of B
  const int arow = t >> 1;              // 0..127
  const int akc  = (t & 1) * 16;
  const int brow = t >> 1;              // 0..127 (n index)
  const int bkc  = (t & 1) * 16;

  v8f acc[4][2];
#pragma unroll
  for (int i = 0; i < 4; ++i)
#pragma unroll
    for (int j = 0; j < 2; ++j) acc[i][j] = vzero8();

  // ---- prologue: stage K-step 0 into buffer 0 ----
  {
    const __bf16* ap = A + (size_t)(m0 + arow) * K + akc;
    v8bf a0 = *(const v8bf*)(ap);
    v8bf a1 = *(const v8bf*)(ap + 8);
    *(v8bf*)(As[0] + arow * 40 + akc)     = a0;
    *(v8bf*)(As[0] + arow * 40 + akc + 8) = a1;
    const __bf16* bp = B + (size_t)(n0 + brow) * K + bkc;
    v8bf b0 = *(const v8bf*)(bp);
    v8bf b1 = *(const v8bf*)(bp + 8);
#pragma unroll
    for (int i = 0; i < 8; ++i) {
      Bt[0][(bkc + i) * 136 + brow]     = b0[i];
      Bt[0][(bkc + 8 + i) * 136 + brow] = b1[i];
    }
  }
  __syncthreads();

  const int nsteps = K / 32;
  for (int s = 0; s < nsteps; ++s) {
    const int  cur      = s & 1;
    const bool has_next = (s + 1 < nsteps);

    // ---- issue next tile's global loads early (overlap with WMMA) ----
    v8bf na0, na1, nb0, nb1;
    if (has_next) {
      const int k0 = (s + 1) * 32;
      const __bf16* ap = A + (size_t)(m0 + arow) * K + k0 + akc;
      na0 = *(const v8bf*)(ap);
      na1 = *(const v8bf*)(ap + 8);
      const __bf16* bp = B + (size_t)(n0 + brow) * K + k0 + bkc;
      nb0 = *(const v8bf*)(bp);
      nb1 = *(const v8bf*)(bp + 8);
    }

    // ---- fragments (ISA 7.12.2 layouts) + 8 WMMA ----
    v16bf af[4], bf2[2];
#pragma unroll
    for (int mf = 0; mf < 4; ++mf) {
      const int row = wm * 64 + mf * 16 + l16;
      v8bf lo = *(const v8bf*)(As[cur] + row * 40 + half * 8);
      v8bf hi = *(const v8bf*)(As[cur] + row * 40 + 16 + half * 8);
      af[mf] = pack16(lo, hi);
    }
#pragma unroll
    for (int nf = 0; nf < 2; ++nf) {
      const int nb = wn * 32 + nf * 16 + l16;
      v16bf bv;
#pragma unroll
      for (int i = 0; i < 8; ++i) {
        bv[i]     = Bt[cur][(half * 8 + i) * 136 + nb];
        bv[i + 8] = Bt[cur][(16 + half * 8 + i) * 136 + nb];
      }
      bf2[nf] = bv;
    }
#pragma unroll
    for (int mf = 0; mf < 4; ++mf)
#pragma unroll
      for (int nf = 0; nf < 2; ++nf)
        acc[mf][nf] = wmma_bf16(af[mf], bf2[nf], acc[mf][nf]);

    // ---- store next tile into the other buffer ----
    if (has_next) {
      const int nxt = cur ^ 1;
      *(v8bf*)(As[nxt] + arow * 40 + akc)     = na0;
      *(v8bf*)(As[nxt] + arow * 40 + akc + 8) = na1;
#pragma unroll
      for (int i = 0; i < 8; ++i) {
        Bt[nxt][(bkc + i) * 136 + brow]     = nb0[i];
        Bt[nxt][(bkc + 8 + i) * 136 + brow] = nb1[i];
      }
    }
    __syncthreads();
  }

  // ---- epilogue: C layout VGPR r -> row r + 8*half, col lane%16 ----
#pragma unroll
  for (int mf = 0; mf < 4; ++mf)
#pragma unroll
    for (int nf = 0; nf < 2; ++nf)
#pragma unroll
      for (int r = 0; r < 8; ++r) {
        const int row = m0 + wm * 64 + mf * 16 + r + half * 8;
        const int col = n0 + wn * 32 + nf * 16 + l16;
        const float v = acc[mf][nf][r];
        if (OUT_BF16) Cb[(size_t)row * N + col] = (__bf16)v;
        else          Cf[(size_t)row * N + col] = v;
      }
}

// ---------------------------------------------------------------------------
// Flash attention (causal).  qkv layout: [b][s][3][h][dh] bf16.
// Block: 128 threads = 4 waves; block handles one 64-row q tile of one
// (batch, head). Each wave owns 16 q rows x full DHEAD=64.
// Softmax scale (1/8, exact in bf16) is folded into the Q fragments.
// ---------------------------------------------------------------------------
__global__ __launch_bounds__(128)
void flash_attn(const __bf16* __restrict__ qkv, __bf16* __restrict__ out) {
  __shared__ __bf16 Kt[64 * 72];        // [d][kv]  (K transposed), pitch 72
  __shared__ __bf16 Vs[64 * 72];        // [kv][d]  natural, pitch 72
  __shared__ __bf16 Ps[4 * 16 * 72];    // per-wave P tile 16x64, pitch 72

  const int t    = threadIdx.x;
  const int wid  = t >> 5;
  const int lane = t & 31;
  const int half = lane >> 4;
  const int l16  = lane & 15;
  const int qt   = blockIdx.x;
  const int h    = blockIdx.y;
  const int b    = blockIdx.z;
  const int q0   = qt * 64;

  const size_t base = (size_t)b * SEQ * 3072;

  // ---- Q fragments, pre-scaled by 1/sqrt(DHEAD) = 0.125 (exact in bf16) ---
  const int qrow = q0 + wid * 16 + l16;
  const __bf16* qptr = qkv + base + (size_t)qrow * 3072 + h * 64;
  v16bf qf[2];
#pragma unroll
  for (int c = 0; c < 2; ++c) {
    v8bf lo = *(const v8bf*)(qptr + c * 32 + half * 8);
    v8bf hi = *(const v8bf*)(qptr + c * 32 + 16 + half * 8);
    v16bf v;
#pragma unroll
    for (int i = 0; i < 8; ++i) {
      v[i]     = (__bf16)((float)lo[i] * 0.125f);
      v[i + 8] = (__bf16)((float)hi[i] * 0.125f);
    }
    qf[c] = v;
  }

  v8f o[4];
#pragma unroll
  for (int i = 0; i < 4; ++i) o[i] = vzero8();
  float mst[8], lst[8];
#pragma unroll
  for (int r = 0; r < 8; ++r) { mst[r] = -3.0e38f; lst[r] = 0.0f; }

  __bf16* Pw = Ps + wid * 16 * 72;

  const int kvr = t >> 1;
  const int dc  = (t & 1) * 32;

  for (int kt = 0; kt <= qt; ++kt) {
    const int kv0 = kt * 64;
    const __bf16* kp = qkv + base + (size_t)(kv0 + kvr) * 3072 + 1024 + h * 64 + dc;
    const __bf16* vp = qkv + base + (size_t)(kv0 + kvr) * 3072 + 2048 + h * 64 + dc;

    // prefetch next tile while this one is processed
    if (kt < qt) {
      __builtin_prefetch(kp + 64 * 3072, 0, 1);
      __builtin_prefetch(vp + 64 * 3072, 0, 1);
    }

    // ---- stage K (transposed scatter) and V (async / vector) tiles ----
    {
      v8bf kv4[4];
#pragma unroll
      for (int i = 0; i < 4; ++i) kv4[i] = *(const v8bf*)(kp + i * 8);
#pragma unroll
      for (int i = 0; i < 4; ++i)
#pragma unroll
        for (int j = 0; j < 8; ++j)
          Kt[(dc + i * 8 + j) * 72 + kvr] = kv4[i][j];
#if USE_ASYNC_LDS
#pragma unroll
      for (int i = 0; i < 4; ++i)
        __builtin_amdgcn_global_load_async_to_lds_b128(
            (v4i*)(vp + i * 8), (v4i*)(Vs + kvr * 72 + dc + i * 8), 0, 0);
      __builtin_amdgcn_s_wait_asynccnt(0);
#else
#pragma unroll
      for (int i = 0; i < 4; ++i)
        *(v8bf*)(Vs + kvr * 72 + dc + i * 8) = *(const v8bf*)(vp + i * 8);
#endif
    }
    __syncthreads();

    // ---- S = Q * K^T (16x64 per wave), K-dim = DHEAD via 2 chained WMMA ---
    v8f s[4];
#pragma unroll
    for (int nf = 0; nf < 4; ++nf) {
      v8f z = vzero8();
#pragma unroll
      for (int c = 0; c < 2; ++c) {
        v16bf kb;
#pragma unroll
        for (int i = 0; i < 8; ++i) {
          kb[i]     = Kt[(c * 32 + half * 8 + i) * 72 + nf * 16 + l16];
          kb[i + 8] = Kt[(c * 32 + 16 + half * 8 + i) * 72 + nf * 16 + l16];
        }
        z = wmma_bf16(qf[c], kb, z);
      }
      s[nf] = z;
    }

    // ---- causal mask (diagonal tile only; scale already folded into Q) ----
    if (kt == qt) {
#pragma unroll
      for (int nf = 0; nf < 4; ++nf)
#pragma unroll
        for (int r = 0; r < 8; ++r) {
          const int rg = q0 + wid * 16 + r + half * 8;
          const int cg = kv0 + nf * 16 + l16;
          if (cg > rg) s[nf][r] = -3.0e38f;
        }
    }

    // ---- online softmax (row stats via 16-lane half-wave reductions) ----
#pragma unroll
    for (int r = 0; r < 8; ++r) {
      float rm = fmaxf(fmaxf(s[0][r], s[1][r]), fmaxf(s[2][r], s[3][r]));
#pragma unroll
      for (int off = 8; off >= 1; off >>= 1)
        rm = fmaxf(rm, __shfl_xor(rm, off, 32));
      const float mnew = fmaxf(mst[r], rm);
      const float fac  = __expf(mst[r] - mnew);
      float rs = 0.0f;
#pragma unroll
      for (int nf = 0; nf < 4; ++nf) {
        const float p = __expf(s[nf][r] - mnew);
        s[nf][r] = p;
        rs += p;
      }
#pragma unroll
      for (int off = 8; off >= 1; off >>= 1)
        rs += __shfl_xor(rs, off, 32);
      lst[r] = lst[r] * fac + rs;
      mst[r] = mnew;
#pragma unroll
      for (int df = 0; df < 4; ++df) o[df][r] *= fac;
    }

    // ---- stage P (C layout -> LDS) so it can be re-read in A layout ----
#pragma unroll
    for (int nf = 0; nf < 4; ++nf)
#pragma unroll
      for (int r = 0; r < 8; ++r)
        Pw[(r + half * 8) * 72 + nf * 16 + l16] = (__bf16)s[nf][r];
    __syncthreads();

    // ---- O += P * V (K-dim = 64 kv rows via 2 chained WMMA) ----
#pragma unroll
    for (int c = 0; c < 2; ++c) {
      v8bf plo = *(const v8bf*)(Pw + l16 * 72 + c * 32 + half * 8);
      v8bf phi = *(const v8bf*)(Pw + l16 * 72 + c * 32 + 16 + half * 8);
      const v16bf pf = pack16(plo, phi);
#pragma unroll
      for (int df = 0; df < 4; ++df) {
        v16bf vb;
#pragma unroll
        for (int i = 0; i < 8; ++i) {
          vb[i]     = Vs[(c * 32 + half * 8 + i) * 72 + df * 16 + l16];
          vb[i + 8] = Vs[(c * 32 + 16 + half * 8 + i) * 72 + df * 16 + l16];
        }
        o[df] = wmma_bf16(pf, vb, o[df]);
      }
    }
    __syncthreads();
  }

  // ---- normalize and store attn output bf16: [b][s][h*64+d] ----
#pragma unroll
  for (int r = 0; r < 8; ++r) {
    const float inv = 1.0f / lst[r];
    const int rg = q0 + wid * 16 + r + half * 8;
    __bf16* op = out + ((size_t)b * SEQ + rg) * DMODEL + h * 64 + l16;
#pragma unroll
    for (int df = 0; df < 4; ++df)
      op[df * 16] = (__bf16)(o[df][r] * inv);
  }
}

// ---------------------------------------------------------------------------
// launch
// ---------------------------------------------------------------------------
extern "C" void kernel_launch(void* const* d_in, const int* in_sizes, int n_in,
                              void* d_out, int out_size, void* d_ws, size_t ws_size,
                              hipStream_t stream) {
  (void)in_sizes; (void)n_in; (void)out_size; (void)ws_size;

  const float* x    = (const float*)d_in[0];
  // d_in[1] is the causal mask; causality is applied structurally.
  const float* Wqkv = (const float*)d_in[2];
  const float* Wo   = (const float*)d_in[3];
  float* out = (float*)d_out;

  char* ws = (char*)d_ws;
  __bf16* xbf   = (__bf16*)ws;  ws += (size_t)ROWS * DMODEL * 2;           //  8 MB
  __bf16* wqkvb = (__bf16*)ws;  ws += (size_t)3 * DMODEL * DMODEL * 2;     //  6 MB
  __bf16* wob   = (__bf16*)ws;  ws += (size_t)DMODEL * DMODEL * 2;         //  2 MB
  __bf16* qkvb  = (__bf16*)ws;  ws += (size_t)ROWS * 3 * DMODEL * 2;       // 24 MB
  __bf16* attnb = (__bf16*)ws;                                             //  8 MB

  cvt_f32_to_bf16<<<1024, 256, 0, stream>>>(x,    xbf,   ROWS * DMODEL);
  cvt_f32_to_bf16<<<1024, 256, 0, stream>>>(Wqkv, wqkvb, 3 * DMODEL * DMODEL);
  cvt_f32_to_bf16<<<512,  256, 0, stream>>>(Wo,   wob,   DMODEL * DMODEL);

  // QKV projection: [4096,1024] x [3072,1024]^T -> bf16 [4096,3072]
  dim3 g1(3 * DMODEL / 128, ROWS / 128);
  gemm_bf16<true><<<g1, 256, 0, stream>>>(xbf, wqkvb, nullptr, qkvb,
                                          ROWS, 3 * DMODEL, DMODEL);

  // Flash attention
  dim3 g2(SEQ / 64, NHEADS, BATCH);
  flash_attn<<<g2, 128, 0, stream>>>(qkvb, attnb);

  // Output projection: [4096,1024] x [1024,1024]^T -> f32 [4096,1024]
  dim3 g3(DMODEL / 128, ROWS / 128);
  gemm_bf16<false><<<g3, 256, 0, stream>>>(attnb, wob, out, nullptr,
                                           ROWS, DMODEL, DMODEL);
}
